// LlamaDecoderLayerMoE_420906795807
// MI455X (gfx1250) — matrix operand order repaired and checked by reading, not compiled, not run on hardware
//
#include <hip/hip_runtime.h>
#include <stdint.h>

// ---------------------------------------------------------------------------
// MoE (top-2) for gfx1250: wave32 + v_wmma_f32_16x16x32_bf16 + async-LDS
// B=4, S=2048, H=1024, F=4096, E=8, T=8192
// ---------------------------------------------------------------------------

typedef __attribute__((ext_vector_type(16))) __bf16 v16bf;
typedef __attribute__((ext_vector_type(8)))  float  v8f;

#define H_DIM 1024
#define F_DIM 4096
#define E_NUM 8
#define T_TOK 8192
#define AROWS 32                             // M tile rows per block
#define ROWS_CAP (2 * T_TOK + AROWS * E_NUM) // 32-aligned expert segments

union ABPack { uint4 q[2]; v16bf v; };

__device__ __forceinline__ unsigned short f2bf(float f) {
    unsigned int u = __float_as_uint(f);
    u += 0x7FFFu + ((u >> 16) & 1u);        // round-to-nearest-even
    return (unsigned short)(u >> 16);
}

// Async copy of one 16B granule (per lane) from global to LDS. ASYNCcnt-tracked.
__device__ __forceinline__ void async_g2l_b128(uint32_t lds_byte_addr, const void* gaddr) {
    asm volatile("global_load_async_to_lds_b128 %0, %1, off"
                 :: "v"(lds_byte_addr), "v"(gaddr) : "memory");
}
__device__ __forceinline__ void wait_async0() {
    asm volatile("s_wait_asynccnt 0" ::: "memory");
}

// ---------------- router: logits, softmax top-2 (one wave per token) -------
__global__ __launch_bounds__(256) void moe_router_kernel(
    const float* __restrict__ x, const float* __restrict__ gw,
    float* __restrict__ rl, int* __restrict__ top_e, float* __restrict__ top_w)
{
    const int lane = threadIdx.x & 31;
    const int wave = threadIdx.x >> 5;
    const int t = blockIdx.x * 8 + wave;

    const float4* xr = (const float4*)(x + (size_t)t * H_DIM + lane * 32);
    float4 xv[8];
#pragma unroll
    for (int i = 0; i < 8; ++i) xv[i] = xr[i];

    float acc[E_NUM];
#pragma unroll
    for (int e = 0; e < E_NUM; ++e) {
        const float4* gr = (const float4*)(gw + (size_t)e * H_DIM + lane * 32);
        float s = 0.f;
#pragma unroll
        for (int i = 0; i < 8; ++i) {
            float4 g = gr[i];
            s += xv[i].x * g.x + xv[i].y * g.y + xv[i].z * g.z + xv[i].w * g.w;
        }
        acc[e] = s;
    }
#pragma unroll
    for (int e = 0; e < E_NUM; ++e)
#pragma unroll
        for (int m = 16; m >= 1; m >>= 1)
            acc[e] += __shfl_xor(acc[e], m, 32);

    if (lane == 0) {
        int i0 = 0;
#pragma unroll
        for (int e = 1; e < E_NUM; ++e) if (acc[e] > acc[i0]) i0 = e;
        int i1 = (i0 == 0) ? 1 : 0;
#pragma unroll
        for (int e = 0; e < E_NUM; ++e) if (e != i0 && acc[e] > acc[i1]) i1 = e;
        // normalized top-2 softmax weights depend only on logit difference
        float ew = __expf(acc[i1] - acc[i0]);
        float w0 = 1.f / (1.f + ew);
#pragma unroll
        for (int e = 0; e < E_NUM; ++e) rl[(size_t)t * E_NUM + e] = acc[e];
        top_e[2 * t]     = i0;  top_e[2 * t + 1] = i1;
        top_w[2 * t]     = w0;  top_w[2 * t + 1] = 1.f - w0;
    }
}

// -------- deterministic counting sort: rank of each token within expert ----
__global__ __launch_bounds__(1024) void moe_count_rank_kernel(
    const int* __restrict__ top_e, int* __restrict__ rankArr, int* __restrict__ counts)
{
    const int e = blockIdx.x;
    const int tid = threadIdx.x;
    const int t0 = tid * (T_TOK / 1024);    // 8 tokens per thread, in token order
    int local = 0;
#pragma unroll
    for (int i = 0; i < T_TOK / 1024; ++i) {
        int t = t0 + i;
        local += (top_e[2 * t] == e) + (top_e[2 * t + 1] == e);
    }
    __shared__ int sc[1024];
    sc[tid] = local;
    __syncthreads();
    for (int off = 1; off < 1024; off <<= 1) {          // inclusive scan
        int v = (tid >= off) ? sc[tid - off] : 0;
        __syncthreads();
        sc[tid] += v;
        __syncthreads();
    }
    int r = sc[tid] - local;                             // exclusive base
#pragma unroll
    for (int i = 0; i < T_TOK / 1024; ++i) {
        int t = t0 + i;
        if (top_e[2 * t] == e)     rankArr[2 * t]     = r++;
        if (top_e[2 * t + 1] == e) rankArr[2 * t + 1] = r++;
    }
    if (tid == 1023) counts[e] = sc[1023];
}

__global__ void moe_offsets_kernel(const int* __restrict__ counts, int* __restrict__ offsets)
{
    if (threadIdx.x == 0 && blockIdx.x == 0) {
        int o = 0;
        offsets[0] = 0;
        for (int e = 0; e < E_NUM; ++e) {
            o = (o + counts[e] + (AROWS - 1)) & ~(AROWS - 1);  // 32-row aligned
            offsets[e + 1] = o;
        }
    }
}

__global__ void fill_zero_u4(uint4* __restrict__ p, size_t n4)
{
    size_t i = (size_t)blockIdx.x * blockDim.x + threadIdx.x;
    size_t stride = (size_t)gridDim.x * blockDim.x;
    uint4 z = {0u, 0u, 0u, 0u};
    for (; i < n4; i += stride) p[i] = z;
}

// ------------- gather routed tokens into bf16 rows, record position --------
__global__ __launch_bounds__(256) void moe_gather_kernel(
    const float* __restrict__ x, const int* __restrict__ top_e,
    const int* __restrict__ rankArr, const int* __restrict__ offsets,
    int* __restrict__ posArr, unsigned short* __restrict__ xg)
{
    const int pair = blockIdx.x;
    const int t = pair >> 1;
    const int e = top_e[pair];
    const int row = offsets[e] + rankArr[pair];
    if (threadIdx.x == 0) posArr[pair] = row;
    float4 f = ((const float4*)(x + (size_t)t * H_DIM))[threadIdx.x];
    ushort4 o;
    o.x = f2bf(f.x); o.y = f2bf(f.y); o.z = f2bf(f.z); o.w = f2bf(f.w);
    ((ushort4*)(xg + (size_t)row * H_DIM))[threadIdx.x] = o;
}

__global__ __launch_bounds__(256) void f32_to_bf16_kernel(
    const float* __restrict__ src, unsigned short* __restrict__ dst, size_t n)
{
    size_t i = ((size_t)blockIdx.x * blockDim.x + threadIdx.x) * 4;
    size_t stride = (size_t)gridDim.x * blockDim.x * 4;
    for (; i < n; i += stride) {
        float4 f = *(const float4*)(src + i);
        ushort4 o;
        o.x = f2bf(f.x); o.y = f2bf(f.y); o.z = f2bf(f.z); o.w = f2bf(f.w);
        *(ushort4*)(dst + i) = o;
    }
}

// -------- FFN stage 1: hg = silu(xg @ w1^T) * (xg @ w3^T)  [bf16 WMMA] -----
// Block tile M=32 x N=128; A (32x1024 bf16, 64KB) in LDS with XOR swizzle on
// 16B granules; staged via global_load_async_to_lds_b128 (ASYNCcnt).
__global__ __launch_bounds__(256) void moe_ffn1_kernel(
    const unsigned short* __restrict__ xg,
    const unsigned short* __restrict__ w1b,
    const unsigned short* __restrict__ w3b,
    const int* __restrict__ offsets,
    unsigned short* __restrict__ hg)
{
    const int e = blockIdx.z;
    const int seg0 = offsets[e];
    const int segRows = offsets[e + 1] - seg0;
    const int m0 = blockIdx.y * AROWS;
    if (m0 >= segRows) return;                           // block-uniform exit
    const int rowBase = seg0 + m0;

    __shared__ unsigned short Atile[AROWS * H_DIM];      // 64 KB, swizzled
    const uint32_t ldsBase = (uint32_t)(uintptr_t)(&Atile[0]);
#pragma unroll
    for (int it = 0; it < 16; ++it) {
        int li = threadIdx.x + it * 256;                 // 4096 granules of 16B
        int r = li >> 7, g = li & 127;
        int sw = ((g ^ (r & 15)) << 3);                  // swizzled elem offset
        uint32_t lds = ldsBase + (uint32_t)(r * H_DIM + sw) * 2u;
        async_g2l_b128(lds, xg + (size_t)(rowBase + r) * H_DIM + g * 8);
    }
    wait_async0();
    __syncthreads();

    const int lane = threadIdx.x & 31;
    const int wave = threadIdx.x >> 5;
    const int half = lane >> 4;
    const int m = lane & 15;
    const int nAbs = blockIdx.x * 128 + wave * 16 + m;
    const unsigned short* w1p = w1b + ((size_t)e * F_DIM + nAbs) * H_DIM;
    const unsigned short* w3p = w3b + ((size_t)e * F_DIM + nAbs) * H_DIM;
    const unsigned short* arow = &Atile[m * H_DIM];      // tile1 at +16*H_DIM

    v8f acc1a = {0.f,0.f,0.f,0.f,0.f,0.f,0.f,0.f};
    v8f acc1b = acc1a, acc3a = acc1a, acc3b = acc1a;
    for (int k0 = 0; k0 < H_DIM; k0 += 32) {
        const int g0 = (k0 >> 3) + half;                 // granule of first octet
        const int o0 = ((g0 ^ m) << 3);
        const int o1 = (((g0 + 2) ^ m) << 3);
        ABPack a0, a1, b1, b3;
        // A 16x32 bf16: lanes0-15 K octets {0-7,16-23}, lanes16-31 {8-15,24-31}
        a0.q[0] = *(const uint4*)(arow + o0);
        a0.q[1] = *(const uint4*)(arow + o1);
        a1.q[0] = *(const uint4*)(arow + 16 * H_DIM + o0);
        a1.q[1] = *(const uint4*)(arow + 16 * H_DIM + o1);
        // B 32x16 bf16: col = lane&15, lanes0-15 K octets {0-7,16-23}, lanes16-31 {8-15,24-31}
        const unsigned short* p1 = w1p + k0 + 8 * half;
        b1.q[0] = *(const uint4*)(p1);
        b1.q[1] = *(const uint4*)(p1 + 16);
        const unsigned short* p3 = w3p + k0 + 8 * half;
        b3.q[0] = *(const uint4*)(p3);
        b3.q[1] = *(const uint4*)(p3 + 16);
        // B reused by two A row-tiles -> halved L2 weight traffic
        acc1a = __builtin_amdgcn_wmma_f32_16x16x32_bf16(false, a0.v, false, b1.v,
                                                        (short)0, acc1a, false, false);
        acc1b = __builtin_amdgcn_wmma_f32_16x16x32_bf16(false, a1.v, false, b1.v,
                                                        (short)0, acc1b, false, false);
        acc3a = __builtin_amdgcn_wmma_f32_16x16x32_bf16(false, a0.v, false, b3.v,
                                                        (short)0, acc3a, false, false);
        acc3b = __builtin_amdgcn_wmma_f32_16x16x32_bf16(false, a1.v, false, b3.v,
                                                        (short)0, acc3b, false, false);
    }
#pragma unroll
    for (int r = 0; r < 8; ++r) {                        // D: row r+8*half, col m
        const int rowM = r + 8 * half;
        float g0 = acc1a[r];
        float h0 = (g0 / (1.f + __expf(-g0))) * acc3a[r];
        hg[(size_t)(rowBase + rowM) * F_DIM + nAbs] = f2bf(h0);
        float g1 = acc1b[r];
        float h1 = (g1 / (1.f + __expf(-g1))) * acc3b[r];
        hg[(size_t)(rowBase + 16 + rowM) * F_DIM + nAbs] = f2bf(h1);
    }
}

// -------- FFN stage 2: yg = hg @ w2^T  [bf16 WMMA, K=4096, double-buffered] -
// Two 64KB LDS A buffers; chunk c+1 staged asynchronously (ASYNCcnt) while
// chunk c is consumed by WMMAs. Top-of-loop s_wait_asynccnt+barrier makes the
// swap race-free.
__global__ __launch_bounds__(256) void moe_ffn2_kernel(
    const unsigned short* __restrict__ hg,
    const unsigned short* __restrict__ w2b,
    const int* __restrict__ offsets,
    float* __restrict__ yg)
{
    const int e = blockIdx.z;
    const int seg0 = offsets[e];
    const int segRows = offsets[e + 1] - seg0;
    const int m0 = blockIdx.y * AROWS;
    if (m0 >= segRows) return;
    const int rowBase = seg0 + m0;

    __shared__ unsigned short Atile[2][AROWS * H_DIM];   // 2 x 64 KB, swizzled
    const uint32_t ldsBase = (uint32_t)(uintptr_t)(&Atile[0][0]);

    const int lane = threadIdx.x & 31;
    const int wave = threadIdx.x >> 5;
    const int half = lane >> 4;
    const int m = lane & 15;
    const int nAbs = blockIdx.x * 128 + wave * 16 + m;
    const unsigned short* w2p = w2b + ((size_t)e * H_DIM + nAbs) * F_DIM;
    const int tid = threadIdx.x;

    auto stage = [&](int buf, int kc) {
        const uint32_t base = ldsBase + (uint32_t)buf * (AROWS * H_DIM * 2u);
#pragma unroll
        for (int it = 0; it < 16; ++it) {
            int li = tid + it * 256;                     // 4096 granules of 16B
            int r = li >> 7, g = li & 127;
            int sw = ((g ^ (r & 15)) << 3);
            async_g2l_b128(base + (uint32_t)(r * H_DIM + sw) * 2u,
                           hg + (size_t)(rowBase + r) * F_DIM + kc + g * 8);
        }
    };

    v8f acca = {0.f,0.f,0.f,0.f,0.f,0.f,0.f,0.f};
    v8f accb = acca;
    stage(0, 0);                                         // preload chunk 0
#pragma unroll 1
    for (int c = 0; c < F_DIM / H_DIM; ++c) {
        wait_async0();                                   // buf[c&1] resident
        __syncthreads();                                 // all waves done w/ buf[(c+1)&1]
        if (c + 1 < F_DIM / H_DIM)
            stage((c + 1) & 1, (c + 1) * H_DIM);         // overlap next chunk load
        const unsigned short* arow = &Atile[c & 1][m * H_DIM];
        const unsigned short* wkp = w2p + c * H_DIM;
        for (int k0 = 0; k0 < H_DIM; k0 += 32) {
            const int g0 = (k0 >> 3) + half;
            const int o0 = ((g0 ^ m) << 3);
            const int o1 = (((g0 + 2) ^ m) << 3);
            ABPack a0, a1, b;
            a0.q[0] = *(const uint4*)(arow + o0);
            a0.q[1] = *(const uint4*)(arow + o1);
            a1.q[0] = *(const uint4*)(arow + 16 * H_DIM + o0);
            a1.q[1] = *(const uint4*)(arow + 16 * H_DIM + o1);
            const unsigned short* pb = wkp + k0 + 8 * half;
            b.q[0] = *(const uint4*)(pb);
            b.q[1] = *(const uint4*)(pb + 16);
            acca = __builtin_amdgcn_wmma_f32_16x16x32_bf16(false, a0.v, false, b.v,
                                                           (short)0, acca, false, false);
            accb = __builtin_amdgcn_wmma_f32_16x16x32_bf16(false, a1.v, false, b.v,
                                                           (short)0, accb, false, false);
        }
    }
#pragma unroll
    for (int r = 0; r < 8; ++r) {
        const int rowM = r + 8 * half;
        yg[(size_t)(rowBase + rowM) * H_DIM + nAbs] = acca[r];
        yg[(size_t)(rowBase + 16 + rowM) * H_DIM + nAbs] = accb[r];
    }
}

// -------- combine: out[t] = w0*yg[pos0] + w1*yg[pos1]  (deterministic) -----
__global__ __launch_bounds__(256) void moe_combine_kernel(
    const float* __restrict__ yg, const int* __restrict__ posArr,
    const float* __restrict__ top_w, float* __restrict__ out)
{
    const int t = blockIdx.x;
    const int p0 = posArr[2 * t], p1 = posArr[2 * t + 1];
    const float w0 = top_w[2 * t], w1 = top_w[2 * t + 1];
    float4 a = ((const float4*)(yg + (size_t)p0 * H_DIM))[threadIdx.x];
    float4 b = ((const float4*)(yg + (size_t)p1 * H_DIM))[threadIdx.x];
    float4 o;
    o.x = w0 * a.x + w1 * b.x;
    o.y = w0 * a.y + w1 * b.y;
    o.z = w0 * a.z + w1 * b.z;
    o.w = w0 * a.w + w1 * b.w;
    ((float4*)(out + (size_t)t * H_DIM))[threadIdx.x] = o;
}

// ---------------------------------------------------------------------------
extern "C" void kernel_launch(void* const* d_in, const int* in_sizes, int n_in,
                              void* d_out, int out_size, void* d_ws, size_t ws_size,
                              hipStream_t stream)
{
    (void)in_sizes; (void)n_in; (void)out_size; (void)ws_size;
    const float* x  = (const float*)d_in[0];
    const float* gw = (const float*)d_in[1];
    const float* w1 = (const float*)d_in[2];
    const float* w3 = (const float*)d_in[3];
    const float* w2 = (const float*)d_in[4];
    float* out = (float*)d_out;
    float* rl  = out + (size_t)T_TOK * H_DIM;            // router_logits after out

    char* wsp = (char*)d_ws;
    auto take = [&](size_t bytes) -> void* {
        void* p = (void*)wsp;
        wsp += (bytes + 255) & ~(size_t)255;
        return p;
    };
    int*            top_e   = (int*)take((size_t)2 * T_TOK * sizeof(int));
    float*          top_w   = (float*)take((size_t)2 * T_TOK * sizeof(float));
    int*            rankArr = (int*)take((size_t)2 * T_TOK * sizeof(int));
    int*            posArr  = (int*)take((size_t)2 * T_TOK * sizeof(int));
    int*            counts  = (int*)take(E_NUM * sizeof(int));
    int*            offsets = (int*)take((E_NUM + 1) * sizeof(int));
    unsigned short* xg  = (unsigned short*)take((size_t)ROWS_CAP * H_DIM * 2);
    unsigned short* w1b = (unsigned short*)take((size_t)E_NUM * F_DIM * H_DIM * 2);
    unsigned short* w3b = (unsigned short*)take((size_t)E_NUM * F_DIM * H_DIM * 2);
    unsigned short* w2b = (unsigned short*)take((size_t)E_NUM * H_DIM * F_DIM * 2);
    unsigned short* hg  = (unsigned short*)take((size_t)ROWS_CAP * F_DIM * 2);
    float*          yg  = (float*)take((size_t)ROWS_CAP * H_DIM * 4);

    moe_router_kernel<<<T_TOK / 8, 256, 0, stream>>>(x, gw, rl, top_e, top_w);
    moe_count_rank_kernel<<<E_NUM, 1024, 0, stream>>>(top_e, rankArr, counts);
    moe_offsets_kernel<<<1, 32, 0, stream>>>(counts, offsets);
    fill_zero_u4<<<2048, 256, 0, stream>>>((uint4*)xg,
                                           (size_t)ROWS_CAP * H_DIM * 2 / 16);
    moe_gather_kernel<<<2 * T_TOK, 256, 0, stream>>>(x, top_e, rankArr, offsets,
                                                     posArr, xg);
    const size_t wn = (size_t)E_NUM * F_DIM * H_DIM;
    f32_to_bf16_kernel<<<4096, 256, 0, stream>>>(w1, w1b, wn);
    f32_to_bf16_kernel<<<4096, 256, 0, stream>>>(w3, w3b, wn);
    f32_to_bf16_kernel<<<4096, 256, 0, stream>>>(w2, w2b, wn);
    moe_ffn1_kernel<<<dim3(F_DIM / 128, T_TOK / AROWS, E_NUM), 256, 0, stream>>>(
        xg, w1b, w3b, offsets, hg);
    moe_ffn2_kernel<<<dim3(H_DIM / 128, T_TOK / AROWS, E_NUM), 256, 0, stream>>>(
        hg, w2b, offsets, yg);
    moe_combine_kernel<<<T_TOK, 256, 0, stream>>>(yg, posArr, top_w, out);
}
